// ComplexGatingNetwork_48000554500925
// MI455X (gfx1250) — compile-verified, matrix operand order checked by reading, not run on hardware
//
#include <hip/hip_runtime.h>
#include <hip/hip_bf16.h>
#include <math.h>

// ---------------------------------------------------------------------------
// ComplexGatingNetwork for MI455X (gfx1250, wave32, WMMA)
//   tokens M = 8192, features K = 4096 (2048 amp + 2048 phase), experts N = 64
//   Roofline: 134 MB of x traffic @ 23.3 TB/s ~ 5.8 us. Loop is VALU/HBM
//   balanced; B loads software-pipelined a full iteration ahead.
// ---------------------------------------------------------------------------

typedef __attribute__((ext_vector_type(16))) _Float16 v16h;
typedef __attribute__((ext_vector_type(8)))  _Float16 v8h;
typedef __attribute__((ext_vector_type(8)))  float    v8f;
typedef __attribute__((ext_vector_type(4)))  float    v4f;

#define BB 4
#define SS 2048
#define DD 2048
#define EE 64
#define M_TOTAL (BB * SS)                     // 8192 tokens
#define KSPLIT 4                              // K-chunks (waves) per M-tile
#define TOK_PER_BLOCK 32                      // 2 M-tiles of 16 per block
#define NUM_BLOCKS (M_TOTAL / TOK_PER_BLOCK)  // 256 blocks x 8 waves = 2048 waves
#define K_CHUNK (DD / KSPLIT)                 // 512 amp-k per wave -> 16 steps

// ---------------------------------------------------------------------------
// fast atan2 at f16-accuracy (A fragment is quantized to f16 anyway).
// 9th-order odd minimax atan on [0,1] + octant fixups; v_rcp_f32 based.
// ---------------------------------------------------------------------------
__device__ __forceinline__ float fast_atan2f(float y, float x) {
  float ax = __builtin_fabsf(x);
  float ay = __builtin_fabsf(y);
  float mx = __builtin_fmaxf(ax, ay);
  float mn = __builtin_fminf(ax, ay);
  float r  = mn * __builtin_amdgcn_rcpf(mx);      // in [0,1]
  r = (mx == 0.0f) ? 0.0f : r;                    // atan2(0,0) = 0
  float r2 = r * r;
  float p = __builtin_fmaf(r2, 0.0208351f, -0.0851330f);
  p = __builtin_fmaf(r2, p, 0.1801410f);
  p = __builtin_fmaf(r2, p, -0.3302995f);
  p = __builtin_fmaf(r2, p, 0.9998660f);
  float a = p * r;                                // atan(mn/mx) in [0, pi/4]
  a = (ay > ax) ? (1.5707963268f - a) : a;        // octant
  a = (x < 0.0f) ? (3.1415926536f - a) : a;       // quadrant
  return __builtin_copysignf(a, y);
}

// ---------------------------------------------------------------------------
// Fragment bundles for the software pipeline
// ---------------------------------------------------------------------------
struct XSet { v4f r0, r1, r2, r3, i0, i1, i2, i3; };
struct BSet { v16h a0, a1, a2, a3, p0, p1, p2, p3; };

__device__ __forceinline__ XSet load_x(const float* __restrict__ xrRow,
                                       const float* __restrict__ xiRow,
                                       int kk, int kq) {
  XSet s;
  s.r0 = *(const v4f*)(xrRow + kk + kq);
  s.r1 = *(const v4f*)(xrRow + kk + kq + 4);
  s.r2 = *(const v4f*)(xrRow + kk + 16 + kq);
  s.r3 = *(const v4f*)(xrRow + kk + 16 + kq + 4);
  s.i0 = *(const v4f*)(xiRow + kk + kq);
  s.i1 = *(const v4f*)(xiRow + kk + kq + 4);
  s.i2 = *(const v4f*)(xiRow + kk + 16 + kq);
  s.i3 = *(const v4f*)(xiRow + kk + 16 + kq + 4);
  return s;
}

__device__ __forceinline__ v16h cat8(v8h lo, v8h hi) {
  return __builtin_shufflevector(lo, hi, 0, 1, 2, 3, 4, 5, 6, 7,
                                 8, 9, 10, 11, 12, 13, 14, 15);
}

__device__ __forceinline__ v16h load_bfrag(const _Float16* __restrict__ wb,
                                           int kk, int kq) {
  v8h lo = *(const v8h*)(wb + kk + kq);
  v8h hi = *(const v8h*)(wb + kk + 16 + kq);
  return cat8(lo, hi);
}

__device__ __forceinline__ BSet load_b(const _Float16* __restrict__ Wt,
                                       int l16, int kk, int kq) {
  BSet b;
  const _Float16* w0 = Wt + (size_t)(0 * 16 + l16) * 4096;
  const _Float16* w1 = Wt + (size_t)(1 * 16 + l16) * 4096;
  const _Float16* w2 = Wt + (size_t)(2 * 16 + l16) * 4096;
  const _Float16* w3 = Wt + (size_t)(3 * 16 + l16) * 4096;
  b.a0 = load_bfrag(w0, kk, kq);      b.p0 = load_bfrag(w0, DD + kk, kq);
  b.a1 = load_bfrag(w1, kk, kq);      b.p1 = load_bfrag(w1, DD + kk, kq);
  b.a2 = load_bfrag(w2, kk, kq);      b.p2 = load_bfrag(w2, DD + kk, kq);
  b.a3 = load_bfrag(w3, kk, kq);      b.p3 = load_bfrag(w3, DD + kk, kq);
  return b;
}

// ---------------------------------------------------------------------------
// Kernel 1: W [4096][64] f32 -> Wt [64][4096] f16 (expert-major, contiguous K)
// ---------------------------------------------------------------------------
__global__ __launch_bounds__(256) void wt_convert_kernel(
    const float* __restrict__ W, _Float16* __restrict__ Wt) {
  int idx = blockIdx.x * 256 + threadIdx.x;     // 0 .. 262143
  int e = idx >> 12;                            // expert  0..63
  int f = idx & 4095;                           // feature 0..4095
  Wt[(size_t)e * 4096 + f] = (_Float16)W[(size_t)f * EE + e];
}

// ---------------------------------------------------------------------------
// Kernel 2: fused amp/phase -> WMMA GEMM -> top-2 gating
// Block: 256 threads = 8 waves = 2 M-tiles x 4 K-chunks.
// Pipeline: loads for step kk+32 issued before the VALU stretch of step kk.
// ---------------------------------------------------------------------------
__global__ __launch_bounds__(256) void gating_kernel(
    const float* __restrict__ xr, const float* __restrict__ xi,
    const _Float16* __restrict__ Wt, const float* __restrict__ bias,
    float* __restrict__ out_probs, float* __restrict__ out_idx) {
  // partial scores from the 4 K-chunk waves; +1 pad column for banks
  __shared__ float sScores[KSPLIT][TOK_PER_BLOCK][EE + 1];

  const int lane  = threadIdx.x & 31;
  const int wave  = threadIdx.x >> 5;
  const int mtile = wave >> 2;       // 0..1
  const int kpart = wave & 3;        // 0..3 : which K slice
  const int half  = lane >> 4;       // A-layout: 0 -> K{0..7,16..23}, 1 -> K{8..15,24..31}
  const int l16   = lane & 15;
  const int kq    = half * 8;

  const int row = blockIdx.x * TOK_PER_BLOCK + mtile * 16 + l16;
  const float* xrRow = xr + (size_t)row * DD;
  const float* xiRow = xi + (size_t)row * DD;

  const int kbeg = kpart * K_CHUNK;
  const int kend = kbeg + K_CHUNK;

  v8f acc0 = {}, acc1 = {}, acc2 = {}, acc3 = {};

  // ---- pipeline prologue
  XSet xc = load_x(xrRow, xiRow, kbeg, kq);
  BSet bc = load_b(Wt, l16, kbeg, kq);

  for (int kk = kbeg; kk < kend; kk += 32) {
    // ---- issue next step's loads (clamped; no control flow)
    const int kn = (kk + 32 < kend) ? (kk + 32) : kbeg;
    __builtin_prefetch(xrRow + kn + 32 + kq, 0, 1);   // 2 steps ahead into L2
    __builtin_prefetch(xiRow + kn + 32 + kq, 0, 1);
    XSet xn = load_x(xrRow, xiRow, kn, kq);
    BSet bn = load_b(Wt, l16, kn, kq);

    // ---- unpack current x
    float fr[16], fi[16];
#pragma unroll
    for (int j = 0; j < 4; ++j) {
      fr[j]      = xc.r0[j]; fr[j + 4]  = xc.r1[j];
      fr[j + 8]  = xc.r2[j]; fr[j + 12] = xc.r3[j];
      fi[j]      = xc.i0[j]; fi[j + 4]  = xc.i1[j];
      fi[j + 8]  = xc.i2[j]; fi[j + 12] = xc.i3[j];
    }

    // ---- amp/phase computed once, f16-accuracy fast paths (~450 VALU:
    // this stretch is what hides the in-flight loads above)
    v16h aAmp, aPh;
#pragma unroll
    for (int j = 0; j < 16; ++j) {
      float a = __builtin_amdgcn_sqrtf(
          __builtin_fmaf(fr[j], fr[j], fi[j] * fi[j]));
      float p = fast_atan2f(fi[j], fr[j]);
      aAmp[j] = (_Float16)a;
      aPh[j]  = (_Float16)p;
    }

    // ---- 8 WMMAs: amp rows [kk) and phase rows [DD+kk) of W
    acc0 = __builtin_amdgcn_wmma_f32_16x16x32_f16(false, aAmp, false, bc.a0,
                                                  (short)0, acc0, false, false);
    acc1 = __builtin_amdgcn_wmma_f32_16x16x32_f16(false, aAmp, false, bc.a1,
                                                  (short)0, acc1, false, false);
    acc2 = __builtin_amdgcn_wmma_f32_16x16x32_f16(false, aAmp, false, bc.a2,
                                                  (short)0, acc2, false, false);
    acc3 = __builtin_amdgcn_wmma_f32_16x16x32_f16(false, aAmp, false, bc.a3,
                                                  (short)0, acc3, false, false);
    acc0 = __builtin_amdgcn_wmma_f32_16x16x32_f16(false, aPh, false, bc.p0,
                                                  (short)0, acc0, false, false);
    acc1 = __builtin_amdgcn_wmma_f32_16x16x32_f16(false, aPh, false, bc.p1,
                                                  (short)0, acc1, false, false);
    acc2 = __builtin_amdgcn_wmma_f32_16x16x32_f16(false, aPh, false, bc.p2,
                                                  (short)0, acc2, false, false);
    acc3 = __builtin_amdgcn_wmma_f32_16x16x32_f16(false, aPh, false, bc.p3,
                                                  (short)0, acc3, false, false);

    // ---- rotate pipeline registers
    xc = xn;
    bc = bn;
  }

  // ---- spill partial scores to LDS (WMMA C striping -> token rows)
  // C layout: vgpr r, lanes 0-15 -> M=r, lanes 16-31 -> M=8+r; N = l16
#pragma unroll
  for (int r = 0; r < 8; ++r) {
    int m = mtile * 16 + half * 8 + r;
    sScores[kpart][m][0 * 16 + l16] = acc0[r];
    sScores[kpart][m][1 * 16 + l16] = acc1[r];
    sScores[kpart][m][2 * 16 + l16] = acc2[r];
    sScores[kpart][m][3 * 16 + l16] = acc3[r];
  }
  __syncthreads();

  // ---- top-2 gating, one thread per token.
  // softmax+top2+renorm collapses to p0 = 1/(1+exp(s1-s0)), p1 = 1-p0.
  if (threadIdx.x < TOK_PER_BLOCK) {
    int t = threadIdx.x;
    int gtok = blockIdx.x * TOK_PER_BLOCK + t;
    float best = -INFINITY, second = -INFINITY;
    int bi = 0, si = 0;
#pragma unroll 8
    for (int e = 0; e < EE; ++e) {
      float s = sScores[0][t][e] + sScores[1][t][e] + sScores[2][t][e] +
                sScores[3][t][e] + bias[e];
      if (s > best) {
        second = best; si = bi;
        best = s;      bi = e;
      } else if (s > second) {
        second = s; si = e;
      }
    }
    float p0 = 1.0f / (1.0f + __expf(second - best));
    out_probs[gtok * 2 + 0] = p0;
    out_probs[gtok * 2 + 1] = 1.0f - p0;
    out_idx[gtok * 2 + 0] = (float)bi;
    out_idx[gtok * 2 + 1] = (float)si;
  }
}

// ---------------------------------------------------------------------------
extern "C" void kernel_launch(void* const* d_in, const int* in_sizes, int n_in,
                              void* d_out, int out_size, void* d_ws, size_t ws_size,
                              hipStream_t stream) {
  const float* x_real = (const float*)d_in[0];   // [4,2048,2048]
  const float* x_imag = (const float*)d_in[1];   // [4,2048,2048]
  const float* W      = (const float*)d_in[2];   // [4096,64]
  const float* bias   = (const float*)d_in[3];   // [64]

  float* out_probs = (float*)d_out;                       // [8192,2]
  float* out_idx   = (float*)d_out + (size_t)M_TOTAL * 2; // [8192,2] as float

  _Float16* Wt = (_Float16*)d_ws;  // 4096*64 f16 = 512 KB scratch

  wt_convert_kernel<<<(4096 * EE) / 256, 256, 0, stream>>>(W, Wt);

  gating_kernel<<<NUM_BLOCKS, 256, 0, stream>>>(x_real, x_imag, Wt, bias,
                                                out_probs, out_idx);
}
